// NonLocalBlock_28132035789523
// MI455X (gfx1250) — matrix operand order repaired and checked by reading, not compiled, not run on hardware
//
#include <hip/hip_runtime.h>
#include <hip/hip_bf16.h>
#include <math.h>

// ---------------------------------------------------------------------------
// NonLocalBlock for MI455X (gfx1250, wave32, WMMA).
// bf16 WMMA 16x16x32 everywhere, f32 accumulation, attention matrix never
// materialized. Fused attention kernel stages phi/g stripes into
// double-buffered LDS with GLOBAL_LOAD_ASYNC_TO_LDS_B128 (ASYNCcnt path),
// overlapping the async DMA of step i+1 with the WMMAs of step i.
// ---------------------------------------------------------------------------

#define B_  4
#define C_  512
#define IC_ 256
#define N_  4096   // 64*64

typedef __attribute__((ext_vector_type(16))) __bf16 v16bf;
typedef __attribute__((ext_vector_type(8)))  __bf16 v8bf;
typedef __attribute__((ext_vector_type(8)))  float  v8f;

__device__ __forceinline__ v8f zero8() { v8f z = {}; return z; }

__device__ __forceinline__ v8f wmma_bf16(v16bf a, v16bf b, v8f c) {
  return __builtin_amdgcn_wmma_f32_16x16x32_bf16(
      /*neg_a=*/false, a, /*neg_b=*/false, b,
      /*c_mod=*/(short)0, c, /*reuse_a=*/false, /*reuse_b=*/false);
}

// A operand (16x32, rows contiguous over K). Lanes 0-15: row M=lane,
// K=0..7,16..23; lanes 16-31: row M=lane-16, K=8..15,24..31.
__device__ __forceinline__ v16bf load_tile_a(const __bf16* base, int ld,
                                             int r0, int k0, int lane) {
  int r  = lane & 15;
  int kh = (lane >> 4) & 1;
  const __bf16* p = base + (size_t)(r0 + r) * ld + k0 + kh * 8;
  v8bf lo = *(const v8bf*)p;
  v8bf hi = *(const v8bf*)(p + 16);
  union { v16bf v; v8bf h[2]; } u;
  u.h[0] = lo; u.h[1] = hi;
  return u.v;
}

// B operand (32x16, columns contiguous over K in memory). Lanes 0-15 hold
// column N=lane with K=0..7,16..23; lanes 16-31 hold column N=lane-16 with K=8..15,24..31.
__device__ __forceinline__ v16bf load_tile_b(const __bf16* base, int ld,
                                             int c0, int k0, int lane) {
  const __bf16* p = base + (size_t)(c0 + (lane & 15)) * ld + k0 + ((lane >> 4) << 3);
  v8bf lo = *(const v8bf*)p;
  v8bf hi = *(const v8bf*)(p + 16);
  union { v16bf v; v8bf h[2]; } u;
  u.h[0] = lo; u.h[1] = hi;
  return u.v;
}

// Per-lane async DMA: 16 bytes global -> LDS, tracked by ASYNCcnt.
__device__ __forceinline__ void async_load_b128(void* lds_ptr, const __bf16* gptr) {
  unsigned lds_off = (unsigned)(unsigned long long)lds_ptr;  // low 32b = LDS addr
  unsigned long long ga = (unsigned long long)gptr;
  asm volatile("global_load_async_to_lds_b128 %0, %1, off"
               :: "v"(lds_off), "v"(ga) : "memory");
}

__device__ __forceinline__ void wait_async0() {
  asm volatile("s_wait_asynccnt 0x0" ::: "memory");
}

__device__ __forceinline__ void atomicMaxF32(float* addr, float val) {
  if (!(val < 0.f)) atomicMax((int*)addr, __float_as_int(val));
  else              atomicMin((unsigned int*)addr, __float_as_uint(val));
}

// ---------------------------------------------------------------------------
// K0: x [b][c][n] f32  ->  xt [b][n][c] bf16 (LDS 32x32 tile transpose)
// ---------------------------------------------------------------------------
__global__ void k_transpose_x(const float* __restrict__ x, __bf16* __restrict__ xt) {
  __shared__ float tile[32][33];
  int b  = blockIdx.z;
  int c0 = blockIdx.y * 32;
  int n0 = blockIdx.x * 32;
  const float* xp = x + ((size_t)b * C_ + c0) * N_ + n0;
#pragma unroll
  for (int i = 0; i < 4; i++)
    tile[threadIdx.y + 8 * i][threadIdx.x] = xp[(size_t)(threadIdx.y + 8 * i) * N_ + threadIdx.x];
  __syncthreads();
  __bf16* op = xt + ((size_t)b * N_ + n0) * C_ + c0;
#pragma unroll
  for (int i = 0; i < 4; i++)
    op[(size_t)(threadIdx.y + 8 * i) * C_ + threadIdx.x] = (__bf16)tile[threadIdx.x][threadIdx.y + 8 * i];
}

// ---------------------------------------------------------------------------
// K0b: fold BN scale into weights (bf16), BN shift into bias (f32)
// ---------------------------------------------------------------------------
__global__ void k_prep_w(const float* __restrict__ w, const float* __restrict__ gamma,
                         const float* __restrict__ beta, const float* __restrict__ mean,
                         const float* __restrict__ var, __bf16* __restrict__ wbf,
                         float* __restrict__ bias, int Cin) {
  int o = blockIdx.x;
  float sc = gamma[o] * rsqrtf(var[o] + 1e-5f);
  if (threadIdx.x == 0) bias[o] = beta[o] - mean[o] * sc;
  for (int c = threadIdx.x; c < Cin; c += blockDim.x)
    wbf[(size_t)o * Cin + c] = (__bf16)(w[(size_t)o * Cin + c] * sc);
}

// ---------------------------------------------------------------------------
// K1: CBR GEMM. out[o,n] = relu(sum_c w'[o,c]*x[c,n] + bias[o])
//     trans=1 -> store [b][n][o] ; trans=0 -> store [b][o][n]
// ---------------------------------------------------------------------------
__global__ void k_cbr(const __bf16* __restrict__ wbf, const float* __restrict__ bias,
                      const __bf16* __restrict__ xt, __bf16* __restrict__ out,
                      int O, int trans) {
  int lane = threadIdx.x & 31;
  int wid  = blockIdx.x * (blockDim.x >> 5) + (threadIdx.x >> 5);
  int tiles_n = N_ >> 4;
  int per_b   = (O >> 4) * tiles_n;
  int b  = wid / per_b;
  int r  = wid % per_b;
  int o0 = (r / tiles_n) << 4;
  int n0 = (r % tiles_n) << 4;
  const __bf16* Bb = xt + (size_t)b * N_ * C_;
  v8f acc = zero8();
#pragma unroll
  for (int k = 0; k < C_; k += 32) {
    v16bf a  = load_tile_a(wbf, C_, o0, k, lane);
    v16bf bb = load_tile_b(Bb, C_, n0, k, lane);
    acc = wmma_bf16(a, bb, acc);
  }
  int col = lane & 15;
  int mrb = (lane >> 4) << 3;
#pragma unroll
  for (int i = 0; i < 8; i++) {
    int o = o0 + mrb + i;
    int n = n0 + col;
    float v = acc[i] + bias[o];
    v = fmaxf(v, 0.f);
    if (trans) out[((size_t)b * N_ + n) * O + o] = (__bf16)v;
    else       out[((size_t)b * O + o) * N_ + n] = (__bf16)v;
  }
}

// ---------------------------------------------------------------------------
// K2: init column stats
// ---------------------------------------------------------------------------
__global__ void k_init_stats(float* __restrict__ colmax, float* __restrict__ colsum) {
  int i = blockIdx.x * blockDim.x + threadIdx.x;
  if (i < B_ * N_) {
    ((unsigned*)colmax)[i] = 0xFF800000u;  // -inf
    colsum[i] = 0.f;
  }
}

// ---------------------------------------------------------------------------
// K3: column stats over n (softmax axis). phase 0: max; phase 1: sum of exp.
// ---------------------------------------------------------------------------
#define NCH 8
__global__ void k_colstat(const __bf16* __restrict__ theta_all,
                          const __bf16* __restrict__ phi_all,
                          const float* __restrict__ colmax,
                          float* __restrict__ dst, int phase) {
  int lane = threadIdx.x & 31;
  int wid  = blockIdx.x * (blockDim.x >> 5) + (threadIdx.x >> 5);
  int per_b = (N_ >> 4) * NCH;
  int b  = wid / per_b;
  int r  = wid % per_b;
  int m0 = (r / NCH) << 4;
  int ch = r % NCH;
  int nbeg = ch * (N_ / NCH);
  const __bf16* th = theta_all + (size_t)b * N_ * IC_;
  const __bf16* ph = phi_all   + (size_t)b * N_ * IC_;

  v16bf Bph[8];
#pragma unroll
  for (int k = 0; k < 8; k++) Bph[k] = load_tile_b(ph, IC_, m0, k * 32, lane);

  int m = m0 + (lane & 15);
  float mx  = (phase == 1) ? colmax[(size_t)b * N_ + m] : 0.f;
  float red = (phase == 1) ? 0.f : -INFINITY;

  for (int n0 = nbeg; n0 < nbeg + N_ / NCH; n0 += 16) {
    v8f a = zero8();
#pragma unroll
    for (int k = 0; k < 8; k++)
      a = wmma_bf16(load_tile_a(th, IC_, n0, k * 32, lane), Bph[k], a);
    if (phase == 0) {
#pragma unroll
      for (int i = 0; i < 8; i++) red = fmaxf(red, a[i]);
    } else {
#pragma unroll
      for (int i = 0; i < 8; i++) red += __expf(a[i] - mx);
    }
  }
  float o = __shfl_xor(red, 16, 32);  // lanes l and l+16 hold the same column
  red = (phase == 1) ? (red + o) : fmaxf(red, o);
  if (lane < 16) {
    float* p = &dst[(size_t)b * N_ + m];
    if (phase == 1) atomicAdd(p, red);
    else            atomicMaxF32(p, red);
  }
}

// ---------------------------------------------------------------------------
// K4: fused attention. Block = 8 waves covering a 64-row n panel of y.
// Per 32-wide m step:
//   * phi stripe (32 x 256 bf16) and g stripe (256 x 32 bf16) are staged into
//     double-buffered LDS by async DMA (issued one step ahead)
//   * 8 waves compute the 8 distinct 16x16 attn tiles of the 64x32 panel,
//     normalize (colmax/colsum), share probabilities through LDS probS
//   * each wave then contracts its (n-tile, c-half) against gS
// ---------------------------------------------------------------------------
__global__ void __launch_bounds__(256)
k_attn_y(const __bf16* __restrict__ theta_all,
         const __bf16* __restrict__ phi_all,
         const __bf16* __restrict__ g_all,
         const float* __restrict__ colmax,
         const float* __restrict__ colsum,
         __bf16* __restrict__ y_all) {
  __shared__ __bf16 phiS[2][32][264];   // rows m_local over c (+8 pad)
  __shared__ __bf16 gS[2][256][40];     // rows c over m_local (+8 pad)
  __shared__ __bf16 probS[64][40];      // rows n_local over m_local (+8 pad)

  int tid  = threadIdx.x;
  int lane = tid & 31;
  int slot = tid >> 5;          // wave 0..7
  int ntile = slot >> 1;        // 0..3  : n sub-tile
  int half  = slot & 1;         // attn m-half / gemm2 c-half
  int b     = blockIdx.x / (N_ / 64);
  int n0    = (blockIdx.x % (N_ / 64)) * 64 + ntile * 16;

  const __bf16* th = theta_all + (size_t)b * N_ * IC_;
  const __bf16* ph = phi_all   + (size_t)b * N_ * IC_;
  const __bf16* gg = g_all     + (size_t)b * IC_ * N_;
  const float*  cm = colmax + (size_t)b * N_;
  const float*  cs = colsum + (size_t)b * N_;

  // stage one m-stripe (32 columns) into LDS buffer `buf` via async DMA
  auto stage = [&](int buf, int m0) {
#pragma unroll
    for (int j = 0; j < 4; j++) {
      int ch = tid + j * 256;                 // 0..1023
      int mm = ch >> 5, pp = ch & 31;         // phi: 32 rows x 32 b128 chunks
      async_load_b128(&phiS[buf][mm][pp * 8], ph + (size_t)(m0 + mm) * IC_ + pp * 8);
      int cc = ch >> 2, p2 = ch & 3;          // g: 256 rows x 4 b128 chunks
      async_load_b128(&gS[buf][cc][p2 * 8], gg + (size_t)cc * N_ + m0 + p2 * 8);
    }
  };

  v16bf Ath[8];
#pragma unroll
  for (int k = 0; k < 8; k++) Ath[k] = load_tile_a(th, IC_, n0, k * 32, lane);

  v8f acc[8];
#pragma unroll
  for (int i = 0; i < 8; i++) acc[i] = zero8();

  int col = lane & 15;
  int mrb = (lane >> 4) << 3;

  stage(0, 0);
  for (int it = 0; it < N_ / 32; it++) {
    int cur = it & 1;
    int m0  = it * 32;
    wait_async0();        // this wave's chunks of buffer `cur` are in LDS
    __syncthreads();      // everyone's chunks are in; prior reads all done
    if (it + 1 < N_ / 32) stage(cur ^ 1, m0 + 32);  // prefetch next stripe

    // attn tile: rows n0..n0+15, cols m0+half*16 .. +15 (phi from LDS)
    v8f a = zero8();
#pragma unroll
    for (int k = 0; k < 8; k++)
      a = wmma_bf16(Ath[k], load_tile_b(&phiS[cur][0][0], 264, half * 16, k * 32, lane), a);
    int m = m0 + half * 16 + col;
    float mx = cm[m];
    float rs = __frcp_rn(cs[m]);
#pragma unroll
    for (int i = 0; i < 8; i++) {
      float p = __expf(a[i] - mx) * rs;
      probS[ntile * 16 + mrb + i][half * 16 + col] = (__bf16)p;
    }
    __syncthreads();

    // y GEMM: this wave covers (n-tile `ntile`, c-half `half`), k = 32 m's
    v16bf Ae = load_tile_a(&probS[ntile * 16][0], 40, 0, 0, lane);
#pragma unroll
    for (int ct = 0; ct < 8; ct++)
      acc[ct] = wmma_bf16(Ae, load_tile_b(&gS[cur][0][0], 40, half * 128 + ct * 16, 0, lane),
                          acc[ct]);
  }

  __bf16* y = y_all + (size_t)b * N_ * IC_;
#pragma unroll
  for (int ct = 0; ct < 8; ct++)
#pragma unroll
    for (int i = 0; i < 8; i++) {
      int n = n0 + mrb + i;
      int c = half * 128 + ct * 16 + col;
      y[(size_t)n * IC_ + c] = (__bf16)acc[ct][i];
    }
}

// ---------------------------------------------------------------------------
// K5: mask CBR + residual. out[b][o][n] = relu(W'y + bias) + x, f32 out.
// ---------------------------------------------------------------------------
__global__ void k_mask_out(const __bf16* __restrict__ wm, const float* __restrict__ bias,
                           const __bf16* __restrict__ y_all,
                           const float* __restrict__ x, float* __restrict__ out) {
  int lane = threadIdx.x & 31;
  int wid  = blockIdx.x * (blockDim.x >> 5) + (threadIdx.x >> 5);
  int tiles_n = N_ >> 4;
  int per_b   = (C_ >> 4) * tiles_n;
  int b  = wid / per_b;
  int r  = wid % per_b;
  int o0 = (r / tiles_n) << 4;
  int n0 = (r % tiles_n) << 4;
  const __bf16* Y = y_all + (size_t)b * N_ * IC_;
  v8f acc = zero8();
#pragma unroll
  for (int k = 0; k < IC_; k += 32)
    acc = wmma_bf16(load_tile_a(wm, IC_, o0, k, lane),
                    load_tile_b(Y, IC_, n0, k, lane), acc);
  int col = lane & 15;
  int mrb = (lane >> 4) << 3;
#pragma unroll
  for (int i = 0; i < 8; i++) {
    int o = o0 + mrb + i;
    int n = n0 + col;
    size_t idx = ((size_t)b * C_ + o) * N_ + n;
    float v = fmaxf(acc[i] + bias[o], 0.f);
    out[idx] = v + x[idx];
  }
}

// ---------------------------------------------------------------------------
// Workspace layout (bytes)
// ---------------------------------------------------------------------------
constexpr size_t OFF_XT  = 0;                                    // x^T bf16
constexpr size_t OFF_WPH = OFF_XT  + (size_t)B_ * N_ * C_ * 2;
constexpr size_t OFF_WTH = OFF_WPH + (size_t)IC_ * C_ * 2;
constexpr size_t OFF_WG  = OFF_WTH + (size_t)IC_ * C_ * 2;
constexpr size_t OFF_WM  = OFF_WG  + (size_t)IC_ * C_ * 2;
constexpr size_t OFF_BPH = OFF_WM  + (size_t)C_ * IC_ * 2;
constexpr size_t OFF_BTH = OFF_BPH + IC_ * 4;
constexpr size_t OFF_BG  = OFF_BTH + IC_ * 4;
constexpr size_t OFF_BM  = OFF_BG  + IC_ * 4;
constexpr size_t OFF_THT = ((OFF_BM + C_ * 4 + 255) & ~(size_t)255);
constexpr size_t OFF_PHT = OFF_THT + (size_t)B_ * N_ * IC_ * 2;
constexpr size_t OFF_GN  = OFF_PHT + (size_t)B_ * N_ * IC_ * 2;
constexpr size_t OFF_CM  = OFF_GN  + (size_t)B_ * IC_ * N_ * 2;
constexpr size_t OFF_CS  = OFF_CM  + (size_t)B_ * N_ * 4;
constexpr size_t OFF_YT  = OFF_CS  + (size_t)B_ * N_ * 4;

extern "C" void kernel_launch(void* const* d_in, const int* in_sizes, int n_in,
                              void* d_out, int out_size, void* d_ws, size_t ws_size,
                              hipStream_t stream) {
  (void)in_sizes; (void)n_in; (void)out_size; (void)ws_size;
  const float* x = (const float*)d_in[0];
  const float* w_phi = (const float*)d_in[1];
  const float* g_phi = (const float*)d_in[2];
  const float* b_phi = (const float*)d_in[3];
  const float* m_phi = (const float*)d_in[4];
  const float* v_phi = (const float*)d_in[5];
  const float* w_th  = (const float*)d_in[6];
  const float* g_th  = (const float*)d_in[7];
  const float* b_th  = (const float*)d_in[8];
  const float* m_th  = (const float*)d_in[9];
  const float* v_th  = (const float*)d_in[10];
  const float* w_g   = (const float*)d_in[11];
  const float* g_g   = (const float*)d_in[12];
  const float* b_g   = (const float*)d_in[13];
  const float* m_g   = (const float*)d_in[14];
  const float* v_g   = (const float*)d_in[15];
  const float* w_mk  = (const float*)d_in[16];
  const float* g_mk  = (const float*)d_in[17];
  const float* b_mk  = (const float*)d_in[18];
  const float* m_mk  = (const float*)d_in[19];
  const float* v_mk  = (const float*)d_in[20];

  char* ws = (char*)d_ws;
  __bf16* xt   = (__bf16*)(ws + OFF_XT);
  __bf16* wphi = (__bf16*)(ws + OFF_WPH);
  __bf16* wth  = (__bf16*)(ws + OFF_WTH);
  __bf16* wg   = (__bf16*)(ws + OFF_WG);
  __bf16* wmk  = (__bf16*)(ws + OFF_WM);
  float*  bphi = (float*)(ws + OFF_BPH);
  float*  bth  = (float*)(ws + OFF_BTH);
  float*  bg   = (float*)(ws + OFF_BG);
  float*  bmk  = (float*)(ws + OFF_BM);
  __bf16* tht  = (__bf16*)(ws + OFF_THT);   // theta [b][n][ic]
  __bf16* pht  = (__bf16*)(ws + OFF_PHT);   // phi   [b][m][ic]
  __bf16* gn   = (__bf16*)(ws + OFF_GN);    // g     [b][ic][m]
  float*  cmx  = (float*)(ws + OFF_CM);
  float*  csm  = (float*)(ws + OFF_CS);
  __bf16* yt   = (__bf16*)(ws + OFF_YT);    // y     [b][n][ic]
  float*  out  = (float*)d_out;

  // 0) transpose+convert x
  k_transpose_x<<<dim3(N_ / 32, C_ / 32, B_), dim3(32, 8), 0, stream>>>(x, xt);

  // 0b) fold BN into weights/bias
  k_prep_w<<<IC_, 256, 0, stream>>>(w_phi, g_phi, b_phi, m_phi, v_phi, wphi, bphi, C_);
  k_prep_w<<<IC_, 256, 0, stream>>>(w_th,  g_th,  b_th,  m_th,  v_th,  wth,  bth,  C_);
  k_prep_w<<<IC_, 256, 0, stream>>>(w_g,   g_g,   b_g,   m_g,   v_g,   wg,   bg,   C_);
  k_prep_w<<<C_,  256, 0, stream>>>(w_mk,  g_mk,  b_mk,  m_mk,  v_mk,  wmk,  bmk,  IC_);

  // 1) three CBR GEMMs
  int cbr_blocks = (B_ * (IC_ / 16) * (N_ / 16)) / 8;
  k_cbr<<<cbr_blocks, 256, 0, stream>>>(wphi, bphi, xt, pht, IC_, 1);
  k_cbr<<<cbr_blocks, 256, 0, stream>>>(wth,  bth,  xt, tht, IC_, 1);
  k_cbr<<<cbr_blocks, 256, 0, stream>>>(wg,   bg,   xt, gn,  IC_, 0);

  // 2) softmax column statistics over n
  k_init_stats<<<(B_ * N_ + 255) / 256, 256, 0, stream>>>(cmx, csm);
  int stat_blocks = (B_ * (N_ / 16) * NCH) / 8;
  k_colstat<<<stat_blocks, 256, 0, stream>>>(tht, pht, cmx, cmx, 0);
  k_colstat<<<stat_blocks, 256, 0, stream>>>(tht, pht, cmx, csm, 1);

  // 3) fused attention -> y (256 blocks x 8 waves, async-LDS double buffered)
  k_attn_y<<<B_ * (N_ / 64), 256, 0, stream>>>(tht, pht, gn, cmx, csm, yt);

  // 4) mask CBR + residual
  k_mask_out<<<(B_ * (C_ / 16) * (N_ / 16)) / 8, 256, 0, stream>>>(wmk, bmk, yt, x, out);
}
